// Attention_2628519985914
// MI455X (gfx1250) — matrix operand order repaired and checked by reading, not compiled, not run on hardware
//
#include <hip/hip_runtime.h>
#include <hip/hip_bf16.h>
#include <cstdint>

#define DEVINL __device__ __forceinline__

constexpr int Bc = 4, Nc = 1024, DIMc = 512, Hc = 8, DHc = 64;
constexpr int INNERc = Hc * DHc;     // 512
constexpr int BNc = Bc * Nc;         // 4096
constexpr int HBc = Hc * Bc;         // 32
constexpr float LNEPS = 1e-5f;

typedef __attribute__((ext_vector_type(16))) __bf16 v16bf;
typedef __attribute__((ext_vector_type(8)))  float  v8f;

DEVINL __bf16 to_bf(float x) { return (__bf16)x; }

DEVINL v8f wmma_bf16(v16bf a, v16bf b, v8f c) {
  // v_wmma_f32_16x16x32_bf16  D = A(16x32) * B(32x16) + C
  return __builtin_amdgcn_wmma_f32_16x16x32_bf16(false, a, false, b, (short)0, c,
                                                 false, false);
}

// ---- CDNA5 async global->LDS copy (ASYNCcnt path, bypasses VGPRs) ---------
// Builtin signature (from hipcc diagnostic): param0 = v4i in AS(1) (global),
// param1 = v4i in AS(3) (LDS), then imm offset, imm cpol. Generic pointers to
// __shared__ carry the LDS byte offset in addr[31:0] (ISA 10.2), so integer
// casts materialize the address-space-qualified pointers.
#if defined(__has_builtin)
#if __has_builtin(__builtin_amdgcn_global_load_async_to_lds_b128)
#define HAVE_ASYNC_LDS_BUILTIN 1
#endif
#if __has_builtin(__builtin_amdgcn_s_wait_asynccnt)
#define HAVE_ASYNCCNT_BUILTIN 1
#endif
#endif

typedef int v4i_vs __attribute__((vector_size(16)));
typedef __attribute__((address_space(1))) v4i_vs* gptr_v4i;
typedef __attribute__((address_space(3))) v4i_vs* lptr_v4i;

DEVINL void async_g2l_b128(void* lds, const void* g) {
#ifdef HAVE_ASYNC_LDS_BUILTIN
  __builtin_amdgcn_global_load_async_to_lds_b128(
      (gptr_v4i)(uintptr_t)g,
      (lptr_v4i)(uint32_t)(uintptr_t)lds, 0, 0);
#else
  asm volatile("global_load_async_to_lds_b128 %0, %1, off"
               :
               : "v"((uint32_t)(uintptr_t)lds), "v"(g)
               : "memory");
#endif
}

DEVINL void wait_async0() {
#ifdef HAVE_ASYNCCNT_BUILTIN
  __builtin_amdgcn_s_wait_asynccnt(0);
#else
  asm volatile("s_wait_asynccnt 0" ::: "memory");
#endif
}

// ---- WMMA fragment loaders (contiguous k = 16h + e order for A and B, wave32) -------------
// A fragment: 16(M) x 32(K), source row-major [M][K], leading dim ld (elems).
// lane: M = lane%16 ; element e -> K = e + 16*(lane>=16)
DEVINL v16bf load_A(const __bf16* base, int ld, int lane) {
  const int m  = lane & 15;
  const int hk = (lane >> 4) << 4;
  v16bf a;
#pragma unroll
  for (int e = 0; e < 16; ++e) {
    const int kk = e + hk;
    a[e] = base[m * ld + kk];
  }
  return a;
}

// B fragment: 32(K) x 16(N) where source is row-major [N][K] (i.e. X @ W^T).
// lane: N = lane%16 ; element e -> K = 16*(lane>=16) + e  (contiguous 32B read)
DEVINL v16bf load_B_nk(const __bf16* base, int ld, int lane) {
  const int n  = lane & 15;
  const int k0 = (lane >> 4) << 4;
  v16bf b;
#pragma unroll
  for (int e = 0; e < 16; ++e) b[e] = base[n * ld + k0 + e];
  return b;
}

// B fragment: 32(K) x 16(N) where source is row-major [K][N].
DEVINL v16bf load_B_kn(const __bf16* base, int ld, int lane) {
  const int n  = lane & 15;
  const int k0 = (lane >> 4) << 4;
  v16bf b;
#pragma unroll
  for (int e = 0; e < 16; ++e) b[e] = base[(k0 + e) * ld + n];
  return b;
}

// ---- kernel 1: fp32 weights -> bf16 ---------------------------------------
__global__ void k_cvt(const float* __restrict__ wi, const float* __restrict__ wo,
                      __bf16* __restrict__ wib, __bf16* __restrict__ wob) {
  const int i = blockIdx.x * 256 + threadIdx.x;
  wib[i] = to_bf(wi[i]);
  wob[i] = to_bf(wo[i]);
}

// ---- kernel 2: LayerNorm of q/k/v rows -> bf16 ----------------------------
__global__ void k_ln(const float* __restrict__ q, const float* __restrict__ k,
                     const float* __restrict__ v, const float* __restrict__ g,
                     const float* __restrict__ bta, __bf16* __restrict__ xln) {
  const int row = blockIdx.x, sel = blockIdx.y, tid = threadIdx.x;
  const float* src = (sel == 0 ? q : (sel == 1 ? k : v)) + (size_t)row * DIMc;
  __shared__ float red[256];
  const float x0 = src[tid], x1 = src[tid + 256];
  red[tid] = x0 + x1;
  __syncthreads();
  for (int off = 128; off; off >>= 1) {
    if (tid < off) red[tid] += red[tid + off];
    __syncthreads();
  }
  const float mean = red[0] * (1.0f / DIMc);
  __syncthreads();
  const float d0 = x0 - mean, d1 = x1 - mean;
  red[tid] = d0 * d0 + d1 * d1;
  __syncthreads();
  for (int off = 128; off; off >>= 1) {
    if (tid < off) red[tid] += red[tid + off];
    __syncthreads();
  }
  const float inv = rsqrtf(red[0] * (1.0f / DIMc) + LNEPS);
  __bf16* dst = xln + ((size_t)sel * BNc + row) * DIMc;
  dst[tid]       = to_bf(d0 * inv * g[tid] + bta[tid]);
  dst[tid + 256] = to_bf(d1 * inv * g[tid + 256] + bta[tid + 256]);
}

// ---- kernel 3/8: Y = X(4096x512) @ W^T(512x512) via bf16 WMMA -------------
// Double-buffered LDS tiles filled by async global->LDS DMA; one barrier per
// K-step; copy of step t+1 overlaps WMMA of step t.
// mode 0: scatter bf16 result into head layout ((h*B+b)*N+n)*DH+d (3 matrices)
// mode 1: fp32 result + bias -> d_out
__global__ void k_gemm(const __bf16* __restrict__ Xall, const __bf16* __restrict__ Wb,
                       const float* __restrict__ bias, __bf16* __restrict__ Ysc,
                       float* __restrict__ Yf, int mode) {
  const int sel  = blockIdx.z;
  const __bf16* X = Xall + (size_t)sel * BNc * DIMc;
  const int row0 = blockIdx.x * 128, col0 = blockIdx.y * 128;
  const int tid = threadIdx.x, lane = tid & 31, wv = tid >> 5;
  const int wm = wv & 3, wn = wv >> 2;      // wave tile: rows wm*32, cols wn*64

  __shared__ __bf16 As[2][128][40];
  __shared__ __bf16 Bs[2][128][40];

  v8f acc[2][4];
#pragma unroll
  for (int m = 0; m < 2; ++m)
#pragma unroll
    for (int n = 0; n < 4; ++n)
#pragma unroll
      for (int r = 0; r < 8; ++r) acc[m][n][r] = 0.0f;

  const int rr = tid >> 1, c0 = (tid & 1) << 4;       // 16-elem (32B) spans
  const __bf16* gA = X  + (size_t)(row0 + rr) * DIMc + c0;
  const __bf16* gB = Wb + (size_t)(col0 + rr) * DIMc + c0;

  // prologue: async-stage K-step 0 into buffer 0
  async_g2l_b128(&As[0][rr][c0], gA);
  async_g2l_b128(&Bs[0][rr][c0], gB);

  int buf = 0;
  for (int k0 = 0; k0 < DIMc; k0 += 32, buf ^= 1) {
    wait_async0();       // this wave's pending copies (into As/Bs[buf]) done
    __syncthreads();     // all waves' copies published; prior reads retired
    if (k0 + 32 < DIMc) {
      async_g2l_b128(&As[buf ^ 1][rr][c0], gA + k0 + 32);
      async_g2l_b128(&Bs[buf ^ 1][rr][c0], gB + k0 + 32);
    }
    v16bf a[2], b[4];
#pragma unroll
    for (int m = 0; m < 2; ++m)
      a[m] = load_A(&As[buf][wm * 32 + m * 16][0], 40, lane);
#pragma unroll
    for (int n = 0; n < 4; ++n)
      b[n] = load_B_nk(&Bs[buf][wn * 64 + n * 16][0], 40, lane);
#pragma unroll
    for (int m = 0; m < 2; ++m)
#pragma unroll
      for (int n = 0; n < 4; ++n) acc[m][n] = wmma_bf16(a[m], b[n], acc[m][n]);
  }

  const int nn = lane & 15, hi = lane >> 4;
#pragma unroll
  for (int m = 0; m < 2; ++m)
#pragma unroll
    for (int n = 0; n < 4; ++n)
#pragma unroll
      for (int r = 0; r < 8; ++r) {
        const int R = row0 + wm * 32 + m * 16 + r + 8 * hi;
        const int C = col0 + wn * 64 + n * 16 + nn;
        const float val = acc[m][n][r];
        if (mode == 0) {
          const int hh = C >> 6, dd = C & 63;        // C = h*DH + d
          const int bb = R >> 10, ni = R & 1023;     // R = b*N + n
          __bf16* Ym = Ysc + (size_t)sel * BNc * DIMc;
          Ym[(((size_t)(hh * Bc + bb)) * Nc + ni) * DHc + dd] = to_bf(val);
        } else {
          Yf[(size_t)R * DIMc + C] = val + bias[C];
        }
      }
}

// ---- kernel 4: per-row stats of f_q / f_k (norm, mean, var ddof=1) --------
__global__ void k_rowstats(const __bf16* __restrict__ fq, const __bf16* __restrict__ fk,
                           float* qn, float* qm, float* qv,
                           float* kn, float* km, float* kv) {
  const int sel = blockIdx.y, lane = threadIdx.x & 31, wv = threadIdx.x >> 5;
  const int row = blockIdx.x * 8 + wv;                  // 0 .. HB*N-1
  const __bf16* f = (sel == 0 ? fq : fk) + (size_t)row * DHc;
  const float e0 = (float)f[lane], e1 = (float)f[lane + 32];
  float s = e0 + e1, ss = e0 * e0 + e1 * e1;
#pragma unroll
  for (int off = 16; off; off >>= 1) {
    s += __shfl_xor(s, off);
    ss += __shfl_xor(ss, off);
  }
  if (lane == 0) {
    const float m = s * (1.0f / DHc);
    const float var = (ss - DHc * m * m) * (1.0f / (DHc - 1));
    if (sel == 0) { qn[row] = sqrtf(ss); qm[row] = m; qv[row] = var; }
    else          { kn[row] = sqrtf(ss); km[row] = m; kv[row] = var; }
  }
}

// ---- kernel 5: global head means q_global/k_global ------------------------
__global__ void k_global(const __bf16* __restrict__ fq, const __bf16* __restrict__ fk,
                         float* qg, float* kg) {
  const int h = blockIdx.x, sel = blockIdx.y, tid = threadIdx.x;
  const __bf16* f = (sel == 0 ? fq : fk) + (size_t)h * Bc * Nc * DHc;
  float* g = (sel == 0 ? qg : kg) + h * DHc;
  const int d = tid & 63, grp = tid >> 6;
  float s = 0.0f;
  for (int row = grp; row < Bc * Nc; row += 4) s += (float)f[(size_t)row * DHc + d];
  __shared__ float red[256];
  red[tid] = s;
  __syncthreads();
  if (tid < 64)
    g[d] = (red[d] + red[d + 64] + red[d + 128] + red[d + 192]) * (1.0f / (Bc * Nc));
}

// ---- kernel 6: tiny weight MLP (LN + ReLU + softmax) ----------------------
__global__ void k_mlp(const float* __restrict__ qg, const float* __restrict__ kg,
                      const float* __restrict__ W1, const float* __restrict__ b1,
                      const float* __restrict__ lg, const float* __restrict__ lb,
                      const float* __restrict__ W2, const float* __restrict__ b2,
                      float* __restrict__ wts) {
  const int j = threadIdx.x;   // 64 threads
  __shared__ float red[64];
  __shared__ float hdn[64];
  for (int h = 0; h < Hc; ++h) {
    float t = b1[j];
    for (int kk = 0; kk < 2 * DHc; ++kk) {
      const float fv = (kk < DHc) ? qg[h * DHc + kk] : kg[h * DHc + kk - DHc];
      t += fv * W1[j * (2 * DHc) + kk];
    }
    red[j] = t;
    __syncthreads();
    for (int off = 32; off; off >>= 1) {
      if (j < off) red[j] += red[j + off];
      __syncthreads();
    }
    const float mean = red[0] * (1.0f / DHc);
    __syncthreads();
    const float dc = t - mean;
    red[j] = dc * dc;
    __syncthreads();
    for (int off = 32; off; off >>= 1) {
      if (j < off) red[j] += red[j + off];
      __syncthreads();
    }
    const float var = red[0] * (1.0f / DHc);
    __syncthreads();
    const float hv = dc * rsqrtf(var + LNEPS) * lg[j] + lb[j];
    hdn[j] = hv > 0.0f ? hv : 0.0f;
    __syncthreads();
    if (j == 0) {
      float lo[3], mx = -1e30f;
      for (int c = 0; c < 3; ++c) {
        float t2 = b2[c];
        for (int kk = 0; kk < DHc; ++kk) t2 += hdn[kk] * W2[c * DHc + kk];
        lo[c] = t2;
        if (t2 > mx) mx = t2;
      }
      float den = 0.0f;
      for (int c = 0; c < 3; ++c) { lo[c] = expf(lo[c] - mx); den += lo[c]; }
      for (int c = 0; c < 3; ++c) wts[h * 3 + c] = lo[c] / den;
    }
    __syncthreads();
  }
}

// ---- kernel 7: streaming mixed-attention  O = mix(Q K^T) @ V --------------
// grid: (N/128 i-tiles, H*B). block: 256 thr = 8 waves, each wave = 16 i-rows.
// K/V j-tiles double-buffered in LDS via async DMA, overlapped with WMMA+mix.
__global__ void k_attn(const __bf16* __restrict__ fq, const __bf16* __restrict__ fk,
                       const __bf16* __restrict__ fv,
                       const float* __restrict__ qn, const float* __restrict__ qm,
                       const float* __restrict__ qv, const float* __restrict__ kn,
                       const float* __restrict__ km, const float* __restrict__ kv,
                       const float* __restrict__ wts, __bf16* __restrict__ Obf) {
  const int hb = blockIdx.y, h = hb >> 2, bb = hb & 3;
  const int i0 = blockIdx.x * 128;
  const int tid = threadIdx.x, lane = tid & 31, wv = tid >> 5;
  const size_t base = (size_t)hb * Nc * DHc;
  const __bf16* Q = fq + base;
  const __bf16* K = fk + base;
  const __bf16* V = fv + base;
  const float* qn_ = qn + (size_t)hb * Nc;
  const float* qm_ = qm + (size_t)hb * Nc;
  const float* qv_ = qv + (size_t)hb * Nc;
  const float* kn_ = kn + (size_t)hb * Nc;
  const float* km_ = km + (size_t)hb * Nc;
  const float* kv_ = kv + (size_t)hb * Nc;
  const float w0 = wts[h * 3 + 0], w1 = wts[h * 3 + 1], w2 = wts[h * 3 + 2];

  __shared__ __bf16 sK[2][64][72];
  __shared__ __bf16 sV[2][64][72];
  __shared__ __bf16 sS[8][16][72];   // per-wave scratch for C->A relayout

  // Per-wave invariant Q fragments (K = 0..31, 32..63) + row stats in regs.
  const int iw = i0 + wv * 16;
  const v16bf aq0 = load_A(Q + (size_t)iw * DHc + 0, DHc, lane);
  const v16bf aq1 = load_A(Q + (size_t)iw * DHc + 32, DHc, lane);
  const int hi = lane >> 4, nn = lane & 15;
  float rqn[8], rqm[8], rqv[8];
#pragma unroll
  for (int r = 0; r < 8; ++r) {
    const int i = iw + r + 8 * hi;
    rqn[r] = qn_[i]; rqm[r] = qm_[i]; rqv[r] = qv_[i];
  }

  v8f acc[4];
#pragma unroll
  for (int dn = 0; dn < 4; ++dn)
#pragma unroll
    for (int r = 0; r < 8; ++r) acc[dn][r] = 0.0f;

  const int rr = tid >> 2, c0 = (tid & 3) << 4;   // 64x64 tile, 16-elem spans
  // prologue: async-stage j-tile 0 into buffer 0
  async_g2l_b128(&sK[0][rr][c0], K + (size_t)rr * DHc + c0);
  async_g2l_b128(&sV[0][rr][c0], V + (size_t)rr * DHc + c0);

  int buf = 0;
  for (int j0 = 0; j0 < Nc; j0 += 64, buf ^= 1) {
    wait_async0();
    __syncthreads();
    if (j0 + 64 < Nc) {
      async_g2l_b128(&sK[buf ^ 1][rr][c0], K + (size_t)(j0 + 64 + rr) * DHc + c0);
      async_g2l_b128(&sV[buf ^ 1][rr][c0], V + (size_t)(j0 + 64 + rr) * DHc + c0);
    }

    // S = Q K^T for this wave's 16 rows, 64 j-cols; mix; drop to LDS as bf16 A.
#pragma unroll
    for (int jn = 0; jn < 4; ++jn) {
      const v16bf bk0 = load_B_nk(&sK[buf][jn * 16][0], 72, lane);
      const v16bf bk1 = load_B_nk(&sK[buf][jn * 16][32], 72, lane);
      v8f s;
#pragma unroll
      for (int r = 0; r < 8; ++r) s[r] = 0.0f;
      s = wmma_bf16(aq0, bk0, s);
      s = wmma_bf16(aq1, bk1, s);
      const int j = j0 + jn * 16 + nn;
      const float cn = kn_[j], cm = km_[j], cv = kv_[j];
#pragma unroll
      for (int r = 0; r < 8; ++r) {
        const float dots = s[r];
        const float mixed = w0 * dots / (rqn[r] * cn)                 // cosine
                          + w1 * (dots * (1.0f / DHc) - rqm[r] * cm)  // cov
                          + w2 * (rqv[r] * cv * (1.0f / DHc));        // var
        sS[wv][r + 8 * hi][jn * 16 + nn] = to_bf(mixed);
      }
    }

    // O += mixed(16x64) @ V(64x64)
    const v16bf as0 = load_A(&sS[wv][0][0], 72, lane);
    const v16bf as1 = load_A(&sS[wv][0][32], 72, lane);
#pragma unroll
    for (int dn = 0; dn < 4; ++dn) {
      const v16bf bv0 = load_B_kn(&sV[buf][0][dn * 16], 72, lane);
      const v16bf bv1 = load_B_kn(&sV[buf][32][dn * 16], 72, lane);
      acc[dn] = wmma_bf16(as0, bv0, acc[dn]);
      acc[dn] = wmma_bf16(as1, bv1, acc[dn]);
    }
  }

  // Store O in (b, n, h*DH + d) bf16 layout for the output projection.
#pragma unroll
  for (int dn = 0; dn < 4; ++dn)
#pragma unroll
    for (int r = 0; r < 8; ++r) {
      const int i = iw + r + 8 * hi;
      const int c = h * DHc + dn * 16 + nn;
      Obf[((size_t)bb * Nc + i) * INNERc + c] = to_bf(acc[dn][r]);
    }
}

// ---------------------------------------------------------------------------
extern "C" void kernel_launch(void* const* d_in, const int* in_sizes, int n_in,
                              void* d_out, int out_size, void* d_ws, size_t ws_size,
                              hipStream_t stream) {
  (void)in_sizes; (void)n_in; (void)out_size; (void)ws_size;
  const float* q      = (const float*)d_in[0];
  const float* k      = (const float*)d_in[1];
  const float* v      = (const float*)d_in[2];
  const float* ln_g   = (const float*)d_in[3];
  const float* ln_b   = (const float*)d_in[4];
  const float* W_in   = (const float*)d_in[5];
  const float* W_out  = (const float*)d_in[6];
  const float* b_out  = (const float*)d_in[7];
  const float* wp_W1  = (const float*)d_in[8];
  const float* wp_b1  = (const float*)d_in[9];
  const float* wp_lg  = (const float*)d_in[10];
  const float* wp_lb  = (const float*)d_in[11];
  const float* wp_W2  = (const float*)d_in[12];
  const float* wp_b2  = (const float*)d_in[13];
  float* out = (float*)d_out;

  uintptr_t p = (uintptr_t)d_ws;
  auto take = [&](size_t bytes) -> void* {
    void* r = (void*)p;
    p += (bytes + 255) & ~(size_t)255;
    return r;
  };
  __bf16* xln = (__bf16*)take((size_t)3 * BNc * DIMc * 2);   // LN'd q/k/v (bf16)
  __bf16* wib = (__bf16*)take((size_t)DIMc * INNERc * 2);    // W_in  bf16
  __bf16* wob = (__bf16*)take((size_t)DIMc * INNERc * 2);    // W_out bf16
  __bf16* fbf = (__bf16*)take((size_t)3 * HBc * Nc * DHc * 2); // f_q,f_k,f_v
  float* qn  = (float*)take((size_t)HBc * Nc * 4);
  float* qm  = (float*)take((size_t)HBc * Nc * 4);
  float* qvv = (float*)take((size_t)HBc * Nc * 4);
  float* kn  = (float*)take((size_t)HBc * Nc * 4);
  float* km  = (float*)take((size_t)HBc * Nc * 4);
  float* kvv = (float*)take((size_t)HBc * Nc * 4);
  float* qg  = (float*)take((size_t)Hc * DHc * 4);
  float* kg  = (float*)take((size_t)Hc * DHc * 4);
  float* wts = (float*)take((size_t)Hc * 3 * 4);
  __bf16* obf = (__bf16*)take((size_t)BNc * INNERc * 2);     // attention output

  __bf16* fq = fbf;
  __bf16* fk = fbf + (size_t)HBc * Nc * DHc;
  __bf16* fvp = fbf + (size_t)2 * HBc * Nc * DHc;

  k_cvt<<<dim3(DIMc * INNERc / 256), dim3(256), 0, stream>>>(W_in, W_out, wib, wob);
  k_ln<<<dim3(BNc, 3), dim3(256), 0, stream>>>(q, k, v, ln_g, ln_b, xln);
  k_gemm<<<dim3(BNc / 128, DIMc / 128, 3), dim3(256), 0, stream>>>(
      xln, wib, nullptr, fbf, nullptr, 0);
  k_rowstats<<<dim3(HBc * Nc / 8, 2), dim3(256), 0, stream>>>(fq, fk, qn, qm, qvv,
                                                              kn, km, kvv);
  k_global<<<dim3(Hc, 2), dim3(256), 0, stream>>>(fq, fk, qg, kg);
  k_mlp<<<dim3(1), dim3(64), 0, stream>>>(qg, kg, wp_W1, wp_b1, wp_lg, wp_lb,
                                          wp_W2, wp_b2, wts);
  k_attn<<<dim3(Nc / 128, HBc), dim3(256), 0, stream>>>(fq, fk, fvp, qn, qm, qvv,
                                                        kn, km, kvv, wts, obf);
  k_gemm<<<dim3(BNc / 128, DIMc / 128, 1), dim3(256), 0, stream>>>(
      obf, wob, b_out, nullptr, out, 1);
}
